// MoDLayer_43877385895981
// MI455X (gfx1250) — matrix operand order repaired and checked by reading, not compiled, not run on hardware
//
#include <hip/hip_runtime.h>
#include <hip/hip_bf16.h>
#include <math.h>

#define B_   4
#define T_   4096
#define D_   1024
#define H_   16
#define HD_  64
#define FF_  4096
#define KK_  512
#define G_   (B_*KK_)   /* 2048 selected rows total */

typedef __bf16 bf16_t;
typedef __attribute__((ext_vector_type(16))) __bf16 v16bf;
typedef __attribute__((ext_vector_type(8)))  __bf16 v8bf;
typedef __attribute__((ext_vector_type(8)))  float  v8f;

union FragAB { v16bf v; v8bf h[2]; };

__device__ inline v8f vzero8(){
  v8f z;
#pragma unroll
  for (int i=0;i<8;++i) z[i]=0.f;
  return z;
}

__device__ inline v8f wmma_bf(v16bf a, v16bf b, v8f c){
  // D = A(16x32 bf16) x B(32x16 bf16) + C(16x16 f32)
  return __builtin_amdgcn_wmma_f32_16x16x32_bf16(false, a, false, b, (short)0, c, false, false);
}

// ---------------- reductions ----------------
__device__ inline float block_reduce_256(float v){
  __shared__ float s[8];
  int l = threadIdx.x & 31, w = threadIdx.x >> 5;
#pragma unroll
  for (int o=16;o;o>>=1) v += __shfl_xor(v, o, 32);
  if (!l) s[w] = v;
  __syncthreads();
  if (w == 0){
    float t = (l < 8) ? s[l] : 0.f;
#pragma unroll
    for (int o=4;o;o>>=1) t += __shfl_xor(t, o, 32);
    if (!l) s[0] = t;
  }
  __syncthreads();
  float r = s[0];
  __syncthreads();
  return r;
}

// ---------------- small utility kernels ----------------
__global__ void k_init_aux(float* p){
  if (threadIdx.x == 0 && blockIdx.x == 0) *p = 0.f;
}

// logits[b,t] = dot(hidden[b,t,:], w_router[:,0])
__global__ __launch_bounds__(256) void k_router(const float* __restrict__ hidden,
                                                const float* __restrict__ wr,
                                                float* __restrict__ logits){
  int tok = blockIdx.x;
  const float* x = hidden + (size_t)tok * D_;
  float s = 0.f;
#pragma unroll
  for (int i=0;i<4;++i){ int c = threadIdx.x + i*256; s += x[c]*wr[c]; }
  s = block_reduce_256(s);
  if (threadIdx.x == 0) logits[tok] = s;
}

// per-batch: top-k by rank, compact ascending idx, gates, aux loss
__global__ __launch_bounds__(1024) void k_topk(const float* __restrict__ logits,
                                               int* __restrict__ idx,
                                               float* __restrict__ gates,
                                               float* __restrict__ aux_out){
  __shared__ float sl[T_];
  __shared__ int   ssum[1024];
  __shared__ float sred[32];
  int b = blockIdx.x, tid = threadIdx.x;
  const float* lg = logits + (size_t)b * T_;
  for (int i = tid; i < T_; i += 1024) sl[i] = lg[i];
  __syncthreads();

  int flags[4]; int cnt = 0; float myaux = 0.f;
  for (int i=0;i<4;++i){
    int t = tid*4 + i;
    float lt = sl[t];
    int rank = 0;
    for (int j=0;j<T_;++j){
      float lj = sl[j];
      rank += (lj > lt) || (lj == lt && j < t);
    }
    int f = (rank < KK_) ? 1 : 0;
    flags[i] = f; cnt += f;
    myaux += fmaxf(lt, 0.f) - lt*(float)f + log1pf(expf(-fabsf(lt)));
  }
  ssum[tid] = cnt;
  __syncthreads();
  for (int off=1; off<1024; off<<=1){
    int v = (tid >= off) ? ssum[tid-off] : 0;
    __syncthreads();
    ssum[tid] += v;
    __syncthreads();
  }
  int pos = ssum[tid] - cnt;   // exclusive prefix
  for (int i=0;i<4;++i){
    if (flags[i]){
      int t = tid*4 + i;
      idx[b*KK_ + pos]   = t;
      gates[b*KK_ + pos] = 1.f/(1.f + expf(-sl[t]));
      pos++;
    }
  }
  // aux reduction (32 waves)
  int l = tid & 31, w = tid >> 5;
#pragma unroll
  for (int o=16;o;o>>=1) myaux += __shfl_xor(myaux, o, 32);
  if (!l) sred[w] = myaux;
  __syncthreads();
  if (w == 0){
    float t = sred[l];
#pragma unroll
    for (int o=16;o;o>>=1) t += __shfl_xor(t, o, 32);
    if (!l) atomicAdd(aux_out, t * (0.01f / (float)(B_*T_)));
  }
}

// gather selected rows + RMSNorm1 -> bf16 activations
__global__ __launch_bounds__(256) void k_gather_rms(const float* __restrict__ hidden,
                                                    const int* __restrict__ idx,
                                                    const float* __restrict__ lnw,
                                                    float* __restrict__ x_sel,
                                                    bf16_t* __restrict__ h1){
  int g = blockIdx.x;
  int b = g >> 9;
  int t = idx[g];
  const float* src = hidden + ((size_t)b*T_ + t) * D_;
  float vals[4]; float ss = 0.f;
#pragma unroll
  for (int i=0;i<4;++i){ int c = threadIdx.x + i*256; vals[i] = src[c]; ss += vals[i]*vals[i]; }
  ss = block_reduce_256(ss);
  float sc = rsqrtf(ss * (1.f/(float)D_) + 1e-6f);
#pragma unroll
  for (int i=0;i<4;++i){
    int c = threadIdx.x + i*256;
    x_sel[(size_t)g*D_ + c] = vals[i];
    h1[(size_t)g*D_ + c] = (bf16_t)(vals[i] * sc * lnw[c]);
  }
}

// W[K][N] fp32 -> Wt[N][K] bf16
__global__ void k_convert(const float* __restrict__ W, bf16_t* __restrict__ Wt, int K, int N){
  size_t total = (size_t)K * N;
  for (size_t i = (size_t)blockIdx.x*blockDim.x + threadIdx.x; i < total;
       i += (size_t)gridDim.x*blockDim.x){
    size_t n = i % N, k = i / N;
    Wt[n*(size_t)K + k] = (bf16_t)W[i];
  }
}

// ---------------- WMMA GEMM: C[M,N] = A[M,K](bf16) @ Wt[N,K]^T ----------------
__global__ __launch_bounds__(128) void k_gemm(const bf16_t* __restrict__ A,
                                              const bf16_t* __restrict__ Wt,
                                              float* __restrict__ C,
                                              int M, int N, int K){
  const int w  = threadIdx.x >> 5;
  const int l  = threadIdx.x & 31;
  const int M0 = blockIdx.y*64 + (w>>1)*32;
  const int N0 = blockIdx.x*64 + (w&1)*32;
  const int lm = l & 15;
  const int hi = l >> 4;
  const int abase = hi*8;
  const int bbase = hi*8;
  const bf16_t* A0 = A  + (size_t)(M0 + lm)*K;
  const bf16_t* A1 = A0 + (size_t)16*K;
  const bf16_t* B0 = Wt + (size_t)(N0 + lm)*K;
  const bf16_t* B1 = B0 + (size_t)16*K;
  v8f acc00=vzero8(), acc01=vzero8(), acc10=vzero8(), acc11=vzero8();
  for (int k=0;k<K;k+=32){
    if ((k & 255) == 0){
      __builtin_prefetch(A0 + k + 512, 0, 1);
      __builtin_prefetch(B0 + k + 512, 0, 1);
    }
    FragAB a0,a1,b0,b1;
    a0.h[0] = *(const v8bf*)(A0 + k + abase);
    a0.h[1] = *(const v8bf*)(A0 + k + abase + 16);
    a1.h[0] = *(const v8bf*)(A1 + k + abase);
    a1.h[1] = *(const v8bf*)(A1 + k + abase + 16);
    b0.h[0] = *(const v8bf*)(B0 + k + bbase);
    b0.h[1] = *(const v8bf*)(B0 + k + bbase + 16);
    b1.h[0] = *(const v8bf*)(B1 + k + bbase);
    b1.h[1] = *(const v8bf*)(B1 + k + bbase + 16);
    acc00 = wmma_bf(a0.v, b0.v, acc00);
    acc01 = wmma_bf(a0.v, b1.v, acc01);
    acc10 = wmma_bf(a1.v, b0.v, acc10);
    acc11 = wmma_bf(a1.v, b1.v, acc11);
  }
#pragma unroll
  for (int r=0;r<8;++r){
    int row0 = M0 + r + hi*8;
    int row1 = row0 + 16;
    C[(size_t)row0*N + N0 + lm]      = acc00[r];
    C[(size_t)row0*N + N0 + 16 + lm] = acc01[r];
    C[(size_t)row1*N + N0 + lm]      = acc10[r];
    C[(size_t)row1*N + N0 + 16 + lm] = acc11[r];
  }
}

// same GEMM but fused epilogue: mid = bf16( silu(gate) * acc )
__global__ __launch_bounds__(128) void k_gemm_silu(const bf16_t* __restrict__ A,
                                                   const bf16_t* __restrict__ Wt,
                                                   const float* __restrict__ gatev,
                                                   bf16_t* __restrict__ mid,
                                                   int M, int N, int K){
  const int w  = threadIdx.x >> 5;
  const int l  = threadIdx.x & 31;
  const int M0 = blockIdx.y*64 + (w>>1)*32;
  const int N0 = blockIdx.x*64 + (w&1)*32;
  const int lm = l & 15;
  const int hi = l >> 4;
  const int abase = hi*8;
  const int bbase = hi*8;
  const bf16_t* A0 = A  + (size_t)(M0 + lm)*K;
  const bf16_t* A1 = A0 + (size_t)16*K;
  const bf16_t* B0 = Wt + (size_t)(N0 + lm)*K;
  const bf16_t* B1 = B0 + (size_t)16*K;
  v8f acc00=vzero8(), acc01=vzero8(), acc10=vzero8(), acc11=vzero8();
  for (int k=0;k<K;k+=32){
    FragAB a0,a1,b0,b1;
    a0.h[0] = *(const v8bf*)(A0 + k + abase);
    a0.h[1] = *(const v8bf*)(A0 + k + abase + 16);
    a1.h[0] = *(const v8bf*)(A1 + k + abase);
    a1.h[1] = *(const v8bf*)(A1 + k + abase + 16);
    b0.h[0] = *(const v8bf*)(B0 + k + bbase);
    b0.h[1] = *(const v8bf*)(B0 + k + bbase + 16);
    b1.h[0] = *(const v8bf*)(B1 + k + bbase);
    b1.h[1] = *(const v8bf*)(B1 + k + bbase + 16);
    acc00 = wmma_bf(a0.v, b0.v, acc00);
    acc01 = wmma_bf(a0.v, b1.v, acc01);
    acc10 = wmma_bf(a1.v, b0.v, acc10);
    acc11 = wmma_bf(a1.v, b1.v, acc11);
  }
#pragma unroll
  for (int r=0;r<8;++r){
    int row0 = M0 + r + hi*8;
    int row1 = row0 + 16;
    {
      size_t i0 = (size_t)row0*N + N0 + lm;
      size_t i1 = (size_t)row0*N + N0 + 16 + lm;
      float g0 = gatev[i0], g1 = gatev[i1];
      mid[i0] = (bf16_t)((g0/(1.f+expf(-g0))) * acc00[r]);
      mid[i1] = (bf16_t)((g1/(1.f+expf(-g1))) * acc01[r]);
    }
    {
      size_t i0 = (size_t)row1*N + N0 + lm;
      size_t i1 = (size_t)row1*N + N0 + 16 + lm;
      float g0 = gatev[i0], g1 = gatev[i1];
      mid[i0] = (bf16_t)((g0/(1.f+expf(-g0))) * acc10[r]);
      mid[i1] = (bf16_t)((g1/(1.f+expf(-g1))) * acc11[r]);
    }
  }
}

// RoPE on q,k; layout q_bh/k_bh [b,h][kk][64] bf16 and vt [b,h][64][kk] bf16
__global__ void k_rope(const float* __restrict__ qraw, const float* __restrict__ kraw,
                       const float* __restrict__ vraw, const int* __restrict__ idx,
                       bf16_t* __restrict__ q_bh, bf16_t* __restrict__ k_bh,
                       bf16_t* __restrict__ vt){
  int i = blockIdx.x*blockDim.x + threadIdx.x;
  if (i >= B_*H_*KK_*HD_) return;
  int d = i & 63;
  int p = (i >> 6) & 511;
  int h = (i >> 15) & 15;
  int b = i >> 19;
  int g = b*KK_ + p;
  float pos = (float)idx[g];
  int f = d & 31;
  float inv_freq = powf(10000.f, -((float)(2*f)) / 64.f);
  float ang = pos * inv_freq;
  float c = cosf(ang), s = sinf(ang);
  size_t base = (size_t)g*D_ + h*HD_;
  float qa = qraw[base + d], ka = kraw[base + d];
  float qo, ko;
  if (d < 32){
    float q2 = qraw[base + d + 32], k2 = kraw[base + d + 32];
    qo = qa*c - q2*s; ko = ka*c - k2*s;
  } else {
    float q1 = qraw[base + d - 32], k1 = kraw[base + d - 32];
    qo = qa*c + q1*s; ko = ka*c + k1*s;
  }
  size_t ob = ((size_t)(b*H_ + h)*KK_ + p)*HD_ + d;
  q_bh[ob] = (bf16_t)qo;
  k_bh[ob] = (bf16_t)ko;
  vt[((size_t)(b*H_ + h)*HD_ + d)*KK_ + p] = (bf16_t)vraw[base + d];
}

// attention: per (chunk-of-32-rows, head, batch); 2 waves, each wave 16 query rows.
__global__ __launch_bounds__(64) void k_attn(const bf16_t* __restrict__ q_bh,
                                             const bf16_t* __restrict__ k_bh,
                                             const bf16_t* __restrict__ vt,
                                             bf16_t* __restrict__ o_bf){
  __shared__ float S[2][16][KK_];   // 64 KB
  const int w = threadIdx.x >> 5;
  const int l = threadIdx.x & 31;
  const int chunk = blockIdx.x, h = blockIdx.y, b = blockIdx.z;
  const int bh = b*H_ + h;
  const bf16_t* Q  = q_bh + (size_t)bh*KK_*HD_;
  const bf16_t* Kp = k_bh + (size_t)bh*KK_*HD_;
  const bf16_t* Vt = vt   + (size_t)bh*HD_*KK_;
  const int rowbase = chunk*32 + w*16;
  const int lm = l & 15;
  const int hi = l >> 4;
  const int abase = hi*8;
  const int bbase = hi*8;
  const bf16_t* Arow = Q + (size_t)(rowbase + lm)*HD_;

  // scores: S[w][m][j] = (q . k)/8, causal-masked
  for (int jt=0; jt<KK_/16; ++jt){
    v8f acc = vzero8();
    const bf16_t* Brow = Kp + (size_t)(jt*16 + lm)*HD_;
#pragma unroll
    for (int ks=0; ks<HD_; ks+=32){
      FragAB a, bb;
      a.h[0]  = *(const v8bf*)(Arow + ks + abase);
      a.h[1]  = *(const v8bf*)(Arow + ks + abase + 16);
      bb.h[0] = *(const v8bf*)(Brow + ks + bbase);
      bb.h[1] = *(const v8bf*)(Brow + ks + bbase + 16);
      acc = wmma_bf(a.v, bb.v, acc);
    }
#pragma unroll
    for (int r=0;r<8;++r){
      int m0 = r + hi*8;
      int kc = jt*16 + lm;
      float v = acc[r] * 0.125f;
      if (kc > rowbase + m0) v = -3.0e38f;
      S[w][m0][kc] = v;
    }
  }
  __syncthreads();

  // softmax per row (wave-private band)
  for (int r=0;r<16;++r){
    float* row = &S[w][r][0];
    float mx = -3.0e38f;
    for (int j=l;j<KK_;j+=32) mx = fmaxf(mx, row[j]);
#pragma unroll
    for (int o=16;o;o>>=1) mx = fmaxf(mx, __shfl_xor(mx, o, 32));
    float sm = 0.f;
    for (int j=l;j<KK_;j+=32){ float e = expf(row[j]-mx); row[j] = e; sm += e; }
#pragma unroll
    for (int o=16;o;o>>=1) sm += __shfl_xor(sm, o, 32);
    float inv = 1.f/sm;
    for (int j=l;j<KK_;j+=32) row[j] *= inv;
  }
  __syncthreads();

  // O = P @ V  (K=512, N=64)
  v8f oacc[4];
#pragma unroll
  for (int nt=0;nt<4;++nt) oacc[nt] = vzero8();
  const float* Pr = &S[w][lm][0];
  for (int k0=0; k0<KK_; k0+=32){
    FragAB a;
#pragma unroll
    for (int j=0;j<8;++j){
      a.v[j]   = (bf16_t)Pr[k0 + abase + j];
      a.v[8+j] = (bf16_t)Pr[k0 + abase + 16 + j];
    }
#pragma unroll
    for (int nt=0;nt<4;++nt){
      const bf16_t* Brow = Vt + (size_t)(nt*16 + lm)*KK_ + k0 + bbase;
      FragAB bb;
      bb.h[0] = *(const v8bf*)(Brow);
      bb.h[1] = *(const v8bf*)(Brow + 16);
      oacc[nt] = wmma_bf(a.v, bb.v, oacc[nt]);
    }
  }
#pragma unroll
  for (int nt=0;nt<4;++nt){
#pragma unroll
    for (int r=0;r<8;++r){
      int m0 = r + hi*8;
      int d = nt*16 + lm;
      o_bf[((size_t)(b*KK_) + rowbase + m0)*D_ + h*HD_ + d] = (bf16_t)oacc[nt][r];
    }
  }
}

// x1 = x_sel + attn_out; h2 = bf16(rmsnorm(x1)*ln2)
__global__ __launch_bounds__(256) void k_resid_rms2(const float* __restrict__ x_sel,
                                                    const float* __restrict__ oo,
                                                    const float* __restrict__ lnw,
                                                    float* __restrict__ x1,
                                                    bf16_t* __restrict__ h2){
  int g = blockIdx.x;
  float vals[4]; float ss = 0.f;
#pragma unroll
  for (int i=0;i<4;++i){
    int c = threadIdx.x + i*256;
    vals[i] = x_sel[(size_t)g*D_ + c] + oo[(size_t)g*D_ + c];
    ss += vals[i]*vals[i];
  }
  ss = block_reduce_256(ss);
  float sc = rsqrtf(ss * (1.f/(float)D_) + 1e-6f);
#pragma unroll
  for (int i=0;i<4;++i){
    int c = threadIdx.x + i*256;
    x1[(size_t)g*D_ + c] = vals[i];
    h2[(size_t)g*D_ + c] = (bf16_t)(vals[i] * sc * lnw[c]);
  }
}

// block_out = x1 + down; new = x_sel + gate*(block_out - x_sel); scatter to out
__global__ __launch_bounds__(256) void k_final(const float* __restrict__ x_sel,
                                               const float* __restrict__ x1,
                                               const float* __restrict__ dwn,
                                               const float* __restrict__ gates,
                                               const int* __restrict__ idx,
                                               float* __restrict__ out){
  int g = blockIdx.x;
  int b = g >> 9;
  int t = idx[g];
  float gt = gates[g];
  float* dst = out + ((size_t)b*T_ + t)*D_;
#pragma unroll
  for (int i=0;i<4;++i){
    int c = threadIdx.x + i*256;
    float bo = x1[(size_t)g*D_ + c] + dwn[(size_t)g*D_ + c];
    float xs = x_sel[(size_t)g*D_ + c];
    dst[c] = xs + gt*(bo - xs);
  }
}

// ---------------- host launch ----------------
extern "C" void kernel_launch(void* const* d_in, const int* in_sizes, int n_in,
                              void* d_out, int out_size, void* d_ws, size_t ws_size,
                              hipStream_t stream){
  (void)in_sizes; (void)n_in; (void)out_size; (void)ws_size;
  const float* hidden   = (const float*)d_in[0];
  const float* w_router = (const float*)d_in[1];
  const float* wq  = (const float*)d_in[2];
  const float* wk  = (const float*)d_in[3];
  const float* wv  = (const float*)d_in[4];
  const float* wo  = (const float*)d_in[5];
  const float* ln1 = (const float*)d_in[6];
  const float* ln2 = (const float*)d_in[7];
  const float* wg  = (const float*)d_in[8];
  const float* wu  = (const float*)d_in[9];
  const float* wd  = (const float*)d_in[10];
  float* out = (float*)d_out;

  char* ws = (char*)d_ws;
  size_t off = 0;
  auto take = [&](size_t bytes)->void*{
    void* p = ws + off;
    off += (bytes + 255) & ~(size_t)255;
    return p;
  };
  bf16_t* wqT  = (bf16_t*)take((size_t)D_*D_*2);
  bf16_t* wkT  = (bf16_t*)take((size_t)D_*D_*2);
  bf16_t* wvT  = (bf16_t*)take((size_t)D_*D_*2);
  bf16_t* woT  = (bf16_t*)take((size_t)D_*D_*2);
  bf16_t* wgT  = (bf16_t*)take((size_t)D_*FF_*2);
  bf16_t* wuT  = (bf16_t*)take((size_t)D_*FF_*2);
  bf16_t* wdT  = (bf16_t*)take((size_t)FF_*D_*2);
  float*  logits = (float*)take((size_t)B_*T_*4);
  int*    idx    = (int*)  take((size_t)B_*KK_*4);
  float*  gates  = (float*)take((size_t)B_*KK_*4);
  float*  x_sel  = (float*)take((size_t)G_*D_*4);
  bf16_t* h1     = (bf16_t*)take((size_t)G_*D_*2);
  float*  qraw   = (float*)take((size_t)G_*D_*4);
  float*  kraw   = (float*)take((size_t)G_*D_*4);
  float*  vraw   = (float*)take((size_t)G_*D_*4);
  bf16_t* q_bh   = (bf16_t*)take((size_t)B_*H_*KK_*HD_*2);
  bf16_t* k_bh   = (bf16_t*)take((size_t)B_*H_*KK_*HD_*2);
  bf16_t* vt     = (bf16_t*)take((size_t)B_*H_*KK_*HD_*2);
  bf16_t* o_bf   = (bf16_t*)take((size_t)G_*D_*2);
  float*  oo     = (float*)take((size_t)G_*D_*4);
  float*  x1     = (float*)take((size_t)G_*D_*4);
  bf16_t* h2     = (bf16_t*)take((size_t)G_*D_*2);
  float*  graw   = (float*)take((size_t)G_*FF_*4);
  bf16_t* mid    = (bf16_t*)take((size_t)G_*FF_*2);
  float*  dwn    = (float*)take((size_t)G_*D_*4);

  float* aux = out + (size_t)B_*T_*D_;

  // 1. output starts as a copy of hidden_states
  hipMemcpyAsync(out, hidden, (size_t)B_*T_*D_*sizeof(float),
                 hipMemcpyDeviceToDevice, stream);
  k_init_aux<<<1, 1, 0, stream>>>(aux);

  // 2. router + top-k + gather
  k_router<<<B_*T_, 256, 0, stream>>>(hidden, w_router, logits);
  k_topk<<<B_, 1024, 0, stream>>>(logits, idx, gates, aux);
  k_gather_rms<<<G_, 256, 0, stream>>>(hidden, idx, ln1, x_sel, h1);

  // 3. weight conversion to bf16 transposed
  k_convert<<<2048, 256, 0, stream>>>(wq, wqT, D_, D_);
  k_convert<<<2048, 256, 0, stream>>>(wk, wkT, D_, D_);
  k_convert<<<2048, 256, 0, stream>>>(wv, wvT, D_, D_);
  k_convert<<<2048, 256, 0, stream>>>(wo, woT, D_, D_);
  k_convert<<<2048, 256, 0, stream>>>(wg, wgT, D_, FF_);
  k_convert<<<2048, 256, 0, stream>>>(wu, wuT, D_, FF_);
  k_convert<<<2048, 256, 0, stream>>>(wd, wdT, FF_, D_);

  // 4. QKV projections (WMMA)
  dim3 gq(D_/64, G_/64);
  k_gemm<<<gq, 128, 0, stream>>>(h1, wqT, qraw, G_, D_, D_);
  k_gemm<<<gq, 128, 0, stream>>>(h1, wkT, kraw, G_, D_, D_);
  k_gemm<<<gq, 128, 0, stream>>>(h1, wvT, vraw, G_, D_, D_);

  // 5. RoPE + head layout
  int rt = B_*H_*KK_*HD_;
  k_rope<<<(rt + 255)/256, 256, 0, stream>>>(qraw, kraw, vraw, idx, q_bh, k_bh, vt);

  // 6. attention (WMMA scores + softmax + WMMA PV)
  dim3 ga(KK_/32, H_, B_);
  k_attn<<<ga, 64, 0, stream>>>(q_bh, k_bh, vt, o_bf);

  // 7. output projection + residual + rmsnorm2
  k_gemm<<<gq, 128, 0, stream>>>(o_bf, woT, oo, G_, D_, D_);
  k_resid_rms2<<<G_, 256, 0, stream>>>(x_sel, oo, ln2, x1, h2);

  // 8. MLP: gate, up(fused silu*), down
  dim3 gg(FF_/64, G_/64);
  k_gemm<<<gg, 128, 0, stream>>>(h2, wgT, graw, G_, FF_, D_);
  k_gemm_silu<<<gg, 128, 0, stream>>>(h2, wuT, graw, mid, G_, FF_, D_);
  k_gemm<<<gq, 128, 0, stream>>>(mid, wdT, dwn, G_, D_, FF_);

  // 9. gated residual + scatter into output
  k_final<<<G_, 256, 0, stream>>>(x_sel, x1, dwn, gates, idx, out);
}
